// Multihead_Lima_17282948399122
// MI455X (gfx1250) — matrix operand order repaired and checked by reading, not compiled, not run on hardware
//
#include <hip/hip_runtime.h>

typedef unsigned short u16;
typedef unsigned int   u32;

typedef __attribute__((ext_vector_type(16))) __bf16 v16bf;
typedef __attribute__((ext_vector_type(8)))  __bf16 v8bf;
typedef __attribute__((ext_vector_type(8)))  float  v8f;

#define BSZ   4096
#define DIM   512
#define HID   256
#define HIN   128
#define HOUT  256
#define NH    16
#define IN_F  (NH * (DIM + HIN))   // 10240
#define MIDF  (NH * HID)           // 4096
#define OUTF  (NH * DIM)           // 8192
#define NHHO  (NH * HOUT)          // 4096

#define BM 128
#define BN 128
#define BK 64
#define KP 72                      // LDS row stride in u16 (144B = 9*16B, keeps b128 aligned)

// ---------- bf16 helpers ----------
__device__ __forceinline__ u16 f2bf(float f) {
  u32 u = __float_as_uint(f);
  u += 0x7FFFu + ((u >> 16) & 1u);   // round-to-nearest-even
  return (u16)(u >> 16);
}
__device__ __forceinline__ u32 pack2(float a, float b) {
  return (u32)f2bf(a) | ((u32)f2bf(b) << 16);
}

// ---------- CDNA5 async global->LDS copy (ASYNCcnt path) ----------
// GLOBAL_LOAD_ASYNC_TO_LDS_B128: per-lane 16B global -> per-lane LDS address.
__device__ __forceinline__ void async_b128(void* lds, const void* g) {
  unsigned l = (unsigned)(uintptr_t)lds;            // low 32 bits of flat = LDS offset
  unsigned long long ga = (unsigned long long)(uintptr_t)g;
  asm volatile("global_load_async_to_lds_b128 %0, %1, off"
               :: "v"(l), "v"(ga) : "memory");
}
__device__ __forceinline__ void async_wait() {
  asm volatile("s_wait_asynccnt 0x0" ::: "memory");
}

// ---------- WMMA fragment loads from LDS ----------
// A (16x32 bf16): lane m=l&15, g=l>>4; K = g*8 + {0..7} and g*8+16 + {0..7}
__device__ __forceinline__ v16bf frag_a(const u16* p) {
  v8bf lo = *(const v8bf*)(p);
  v8bf hi = *(const v8bf*)(p + 16);
  return __builtin_shufflevector(lo, hi, 0,1,2,3,4,5,6,7,8,9,10,11,12,13,14,15);
}
// B (32x16 bf16, staged N-major): lane n=l&15, g=l>>4; K = g*8 + {0..7} and g*8+16 + {0..7}
__device__ __forceinline__ v16bf frag_b(const u16* p) {
  v8bf lo = *(const v8bf*)(p);
  v8bf hi = *(const v8bf*)(p + 16);
  return __builtin_shufflevector(lo, hi, 0,1,2,3,4,5,6,7,8,9,10,11,12,13,14,15);
}

// one 32-K WMMA step for a wave's 32x64 sub-tile (2x4 accumulators)
__device__ __forceinline__ void mma_step(const u16* As, const u16* Bs, int ks,
                                         int wm, int wn, int m_lane, int g,
                                         v8f acc[2][4]) {
  v16bf a[2], b[4];
#pragma unroll
  for (int i = 0; i < 2; ++i)
    a[i] = frag_a(As + (wm * 32 + i * 16 + m_lane) * KP + ks * 32 + g * 8);
#pragma unroll
  for (int j = 0; j < 4; ++j)
    b[j] = frag_b(Bs + (wn * 64 + j * 16 + m_lane) * KP + ks * 32 + g * 8);
#pragma unroll
  for (int i = 0; i < 2; ++i)
#pragma unroll
    for (int j = 0; j < 4; ++j)
      acc[i][j] = __builtin_amdgcn_wmma_f32_16x16x32_bf16(
          false, a[i], false, b[j], (short)0, acc[i][j], false, false);
}

// ---------- fp32 [K][N] -> bf16 [N][K] transpose-convert (one-shot per launch) ----------
// 32x32 tiles via LDS; blockIdx.z batches (for per-head Theta).
__global__ __launch_bounds__(256) void cvt_t_bf16(const float* __restrict__ in,
                                                  u16* __restrict__ out, int K, int N) {
  const size_t batch = (size_t)blockIdx.z * (size_t)K * (size_t)N;
  in += batch;
  out += batch;
  __shared__ u16 t[32][33];
  const int k0 = blockIdx.y * 32, n0 = blockIdx.x * 32;
  const int r = threadIdx.x >> 3, cq = threadIdx.x & 7;
  float4 v = *(const float4*)(in + (size_t)(k0 + r) * N + n0 + cq * 4);
  t[r][cq * 4 + 0] = f2bf(v.x);
  t[r][cq * 4 + 1] = f2bf(v.y);
  t[r][cq * 4 + 2] = f2bf(v.z);
  t[r][cq * 4 + 3] = f2bf(v.w);
  __syncthreads();
  const int n = r, kq = cq;
  u32 a = (u32)t[kq * 4 + 0][n] | ((u32)t[kq * 4 + 1][n] << 16);
  u32 b = (u32)t[kq * 4 + 2][n] | ((u32)t[kq * 4 + 3][n] << 16);
  uint2 o;
  o.x = a;
  o.y = b;
  *(uint2*)(out + (size_t)(n0 + n) * K + k0 + kq * 4) = o;
}

// ---------- GEMM1: total_in @ W_in + b_in, fused lima activation, bf16 out ----------
// B operand from pre-transposed W_inT [MIDF][IN_F] via async global->LDS.
__global__ __launch_bounds__(256) void gemm1_lima(
    const float* __restrict__ theta, const float* __restrict__ hist,
    const u16* __restrict__ WbT, const float* __restrict__ b_in,
    const float* __restrict__ prop, const float* __restrict__ lshape,
    u16* __restrict__ o_ws) {
  __shared__ u16 As[BM * KP];
  __shared__ u16 Bs[BN * KP];
  const int tid = threadIdx.x;
  const int lane = tid & 31, wave = tid >> 5;
  const int wm = wave & 3, wn = wave >> 2;
  const int m_lane = lane & 15, g = lane >> 4;
  const int row0 = blockIdx.y * BM;   // batch rows
  const int col0 = blockIdx.x * BN;   // mid cols

  v8f acc[2][4] = {};

  for (int k0 = 0; k0 < IN_F; k0 += BK) {
    // ---- stage B: async DMA rows of W_inT (N-major, K contiguous) ----
#pragma unroll
    for (int jj = 0; jj < 4; ++jj) {
      int idx = tid + jj * 256;            // 0..1023
      int n = idx >> 3;                    // 0..127
      int kq = idx & 7;                    // 8-bf16 chunk
      async_b128(&Bs[n * KP + kq * 8],
                 WbT + (size_t)(col0 + n) * IN_F + k0 + kq * 8);
    }
    // ---- stage A: on-the-fly concat(theta, histories) -> bf16 LDS ----
    {
      const int hh = k0 / (DIM + HIN);
      const int r = k0 - hh * (DIM + HIN); // 64-stage never crosses a segment (640=10*64)
      const float* src;
      int rstride;
      if (r < DIM) { src = theta + (size_t)row0 * DIM + r;                     rstride = DIM; }
      else         { src = hist + ((size_t)row0 * NH + hh) * HIN + (r - DIM);  rstride = NH * HIN; }
#pragma unroll
      for (int jj = 0; jj < 8; ++jj) {
        int idx = tid + jj * 256;          // 0..2047
        int rr = idx >> 4;                 // 0..127
        int kq = idx & 15;                 // float4 column
        float4 v = *(const float4*)(src + (size_t)rr * rstride + kq * 4);
        uint2 p;
        p.x = pack2(v.x, v.y);
        p.y = pack2(v.z, v.w);
        *(uint2*)(&As[rr * KP + kq * 4]) = p;
      }
    }
    async_wait();
    __syncthreads();
#pragma unroll
    for (int ks = 0; ks < 2; ++ks) mma_step(As, Bs, ks, wm, wn, m_lane, g, acc);
    __syncthreads();
  }

  // ---- epilogue: h = acc + b_in; lima(h) -> bf16 ----
#pragma unroll
  for (int j = 0; j < 4; ++j) {
    int c = col0 + wn * 64 + j * 16 + m_lane;          // also flat index into prop/lshape
    float bin = b_in[c];
    float ba = (1.0f / (1.0f + __expf(-lshape[c])) - 0.5f) * 0.5f;
    float scc = 0.5f / (__expf(prop[c]) + 1.0f);       // M=1
#pragma unroll
    for (int i = 0; i < 2; ++i) {
      int rbase = row0 + wm * 32 + i * 16 + g * 8;
#pragma unroll
      for (int r = 0; r < 8; ++r) {
        float hv = acc[i][j][r] + bin;
        float ov = 1.0f + scc * (-1.0f - ba + __cosf(hv) + ba * __cosf(2.0f * hv));
        o_ws[(size_t)(rbase + r) * MIDF + c] = f2bf(ov);
      }
    }
  }
}

// ---------- GEMM2: raw[b,h,:] = o[b,h,:] @ Theta[h]  (batched per head) ----------
// A from o (bf16 row-major), B from ThetaT [h][n][m]; both staged via async DMA.
__global__ __launch_bounds__(256) void gemm2_batched(
    const u16* __restrict__ o_ws, const u16* __restrict__ TbT,
    float* __restrict__ raw_f, u16* __restrict__ raw_b) {
  __shared__ u16 As[BM * KP];
  __shared__ u16 Bs[BN * KP];
  const int tid = threadIdx.x;
  const int lane = tid & 31, wave = tid >> 5;
  const int wm = wave & 3, wn = wave >> 2;
  const int m_lane = lane & 15, g = lane >> 4;
  const int row0 = blockIdx.x * BM;
  const int head = blockIdx.y >> 1;
  const int col0 = (blockIdx.y & 1) * BN;     // within HOUT

  v8f acc[2][4] = {};

  for (int k0 = 0; k0 < HID; k0 += BK) {
#pragma unroll
    for (int jj = 0; jj < 4; ++jj) {
      int idx = tid + jj * 256;
      int rr = idx >> 3;                       // 0..127
      int kq = idx & 7;
      async_b128(&As[rr * KP + kq * 8],
                 o_ws + (size_t)(row0 + rr) * MIDF + head * HID + k0 + kq * 8);
    }
#pragma unroll
    for (int jj = 0; jj < 4; ++jj) {
      int idx = tid + jj * 256;
      int n = idx >> 3;
      int kq = idx & 7;
      async_b128(&Bs[n * KP + kq * 8],
                 TbT + ((size_t)head * HOUT + col0 + n) * HID + k0 + kq * 8);
    }
    async_wait();
    __syncthreads();
#pragma unroll
    for (int ks = 0; ks < 2; ++ks) mma_step(As, Bs, ks, wm, wn, m_lane, g, acc);
    __syncthreads();
  }

#pragma unroll
  for (int j = 0; j < 4; ++j) {
    int cglob = head * HOUT + col0 + wn * 64 + j * 16 + m_lane;
#pragma unroll
    for (int i = 0; i < 2; ++i) {
      int rbase = row0 + wm * 32 + i * 16 + g * 8;
#pragma unroll
      for (int r = 0; r < 8; ++r) {
        float v = acc[i][j][r];
        raw_f[(size_t)(rbase + r) * NHHO + cglob] = v;
        raw_b[(size_t)(rbase + r) * NHHO + cglob] = f2bf(v);
      }
    }
  }
}

// ---------- LayerNorm over last dim (256), one wave per row ----------
__global__ __launch_bounds__(256) void layernorm_k(
    const float* __restrict__ raw_f, const float* __restrict__ ln_w,
    const float* __restrict__ ln_b, float* __restrict__ out2) {
  const int row = blockIdx.x * 8 + (threadIdx.x >> 5);
  const int lane = threadIdx.x & 31;
  const float4* p = (const float4*)(raw_f + (size_t)row * HOUT);
  float4 v0 = p[lane];
  float4 v1 = p[lane + 32];
  float s  = v0.x + v0.y + v0.z + v0.w + v1.x + v1.y + v1.z + v1.w;
  float ss = v0.x * v0.x + v0.y * v0.y + v0.z * v0.z + v0.w * v0.w +
             v1.x * v1.x + v1.y * v1.y + v1.z * v1.z + v1.w * v1.w;
#pragma unroll
  for (int off = 16; off > 0; off >>= 1) {
    s  += __shfl_xor(s, off, 32);
    ss += __shfl_xor(ss, off, 32);
  }
  float mean = s * (1.0f / HOUT);
  float var = ss * (1.0f / HOUT) - mean * mean;
  float rstd = rsqrtf(var + 1e-5f);
  const float4* wv = (const float4*)ln_w;
  const float4* bv = (const float4*)ln_b;
  float4* q = (float4*)(out2 + (size_t)row * HOUT);
  float4 w = wv[lane], bb = bv[lane], o;
  o.x = (v0.x - mean) * rstd * w.x + bb.x;
  o.y = (v0.y - mean) * rstd * w.y + bb.y;
  o.z = (v0.z - mean) * rstd * w.z + bb.z;
  o.w = (v0.w - mean) * rstd * w.w + bb.w;
  q[lane] = o;
  w = wv[lane + 32]; bb = bv[lane + 32];
  o.x = (v1.x - mean) * rstd * w.x + bb.x;
  o.y = (v1.y - mean) * rstd * w.y + bb.y;
  o.z = (v1.z - mean) * rstd * w.z + bb.z;
  o.w = (v1.w - mean) * rstd * w.w + bb.w;
  q[lane + 32] = o;
}

// ---------- GEMM3: raw @ W_out + b_out -> output ----------
// A from raw bf16 row-major, B from W_outT [OUTF][NHHO]; both via async DMA.
__global__ __launch_bounds__(256) void gemm3_out(
    const u16* __restrict__ raw_b, const u16* __restrict__ WobT,
    const float* __restrict__ b_out, float* __restrict__ out1) {
  __shared__ u16 As[BM * KP];
  __shared__ u16 Bs[BN * KP];
  const int tid = threadIdx.x;
  const int lane = tid & 31, wave = tid >> 5;
  const int wm = wave & 3, wn = wave >> 2;
  const int m_lane = lane & 15, g = lane >> 4;
  const int row0 = blockIdx.y * BM;
  const int col0 = blockIdx.x * BN;

  v8f acc[2][4] = {};

  for (int k0 = 0; k0 < NHHO; k0 += BK) {
#pragma unroll
    for (int jj = 0; jj < 4; ++jj) {
      int idx = tid + jj * 256;
      int rr = idx >> 3;
      int kq = idx & 7;
      async_b128(&As[rr * KP + kq * 8],
                 raw_b + (size_t)(row0 + rr) * NHHO + k0 + kq * 8);
    }
#pragma unroll
    for (int jj = 0; jj < 4; ++jj) {
      int idx = tid + jj * 256;
      int n = idx >> 3;
      int kq = idx & 7;
      async_b128(&Bs[n * KP + kq * 8],
                 WobT + (size_t)(col0 + n) * NHHO + k0 + kq * 8);
    }
    async_wait();
    __syncthreads();
#pragma unroll
    for (int ks = 0; ks < 2; ++ks) mma_step(As, Bs, ks, wm, wn, m_lane, g, acc);
    __syncthreads();
  }

#pragma unroll
  for (int j = 0; j < 4; ++j) {
    int c = col0 + wn * 64 + j * 16 + m_lane;
    float bo = b_out[c];
#pragma unroll
    for (int i = 0; i < 2; ++i) {
      int rbase = row0 + wm * 32 + i * 16 + g * 8;
#pragma unroll
      for (int r = 0; r < 8; ++r)
        out1[(size_t)(rbase + r) * OUTF + c] = acc[i][j][r] + bo;
    }
  }
}

extern "C" void kernel_launch(void* const* d_in, const int* in_sizes, int n_in,
                              void* d_out, int out_size, void* d_ws, size_t ws_size,
                              hipStream_t stream) {
  const float* theta  = (const float*)d_in[0];
  const float* hist   = (const float*)d_in[1];
  const float* W_in   = (const float*)d_in[2];
  const float* b_in   = (const float*)d_in[3];
  const float* prop   = (const float*)d_in[4];
  const float* lshape = (const float*)d_in[5];
  const float* Theta  = (const float*)d_in[6];
  const float* ln_w   = (const float*)d_in[7];
  const float* ln_b   = (const float*)d_in[8];
  const float* W_out  = (const float*)d_in[9];
  const float* b_out  = (const float*)d_in[10];

  float* out1 = (float*)d_out;                       // [B, NH, DIM]
  float* out2 = out1 + (size_t)BSZ * NH * DIM;       // [B, NH, HOUT]

  char* ws = (char*)d_ws;
  u16* WinT  = (u16*)ws;  ws += (size_t)IN_F * MIDF * 2;      // 80 MB  [MIDF][IN_F]
  u16* WoutT = (u16*)ws;  ws += (size_t)NHHO * OUTF * 2;      // 64 MB  [OUTF][NHHO]
  u16* ThT   = (u16*)ws;  ws += (size_t)NH * HID * HOUT * 2;  // 2 MB   [h][n][m]
  u16* o_b   = (u16*)ws;  ws += (size_t)BSZ * MIDF * 2;       // 32 MB
  float* rawf = (float*)ws; ws += (size_t)BSZ * NHHO * 4;     // 64 MB
  u16* rawb  = (u16*)ws;                                      // 32 MB

  // one-shot transpose-convert of all weights to bf16 column-major
  cvt_t_bf16<<<dim3(MIDF / 32, IN_F / 32, 1), 256, 0, stream>>>(W_in,  WinT,  IN_F, MIDF);
  cvt_t_bf16<<<dim3(OUTF / 32, NHHO / 32, 1), 256, 0, stream>>>(W_out, WoutT, NHHO, OUTF);
  cvt_t_bf16<<<dim3(HOUT / 32, HID / 32, NH), 256, 0, stream>>>(Theta, ThT,   HID,  HOUT);

  gemm1_lima<<<dim3(MIDF / BN, BSZ / BM), 256, 0, stream>>>(
      theta, hist, WinT, b_in, prop, lshape, o_b);
  gemm2_batched<<<dim3(BSZ / BM, NH * (HOUT / BN)), 256, 0, stream>>>(
      o_b, ThT, rawf, rawb);
  layernorm_k<<<(BSZ * NH) / 8, 256, 0, stream>>>(rawf, ln_w, ln_b, out2);
  gemm3_out<<<dim3(OUTF / BN, BSZ / BM), 256, 0, stream>>>(
      rawb, WoutT, b_out, out1);
}
